// ThermoQuillan_21474836480527
// MI455X (gfx1250) — matrix operand order repaired and checked by reading, not compiled, not run on hardware
//
#include <hip/hip_runtime.h>
#include <hip/hip_bf16.h>

// ---------------------------------------------------------------------------
// ThermoCouncilEBM forward on gfx1250 (MI455X), built from one generic
// bf16-WMMA GEMM kernel:  C = act(A[M,K] @ B[K,N] (+bias) (+C_old))
// Software-pipelined: global loads for chunk k+1 are in flight while the
// v_wmma block for chunk k executes; LDS tiles are double-buffered.
// ---------------------------------------------------------------------------

typedef __attribute__((ext_vector_type(16))) __bf16 v16bf;
typedef __attribute__((ext_vector_type(8)))  __bf16 v8bf;
typedef __attribute__((ext_vector_type(4)))  __bf16 v4bf;
typedef __attribute__((ext_vector_type(8)))  float  v8f;

#define TILE_M 128
#define TILE_N 128
#define TILE_K 32
#define LDSK   40   // 32 + 8 bf16 pad -> 80-byte row stride (16B aligned)

#define FLAG_RELU 1
#define FLAG_ACC  2

static __device__ __forceinline__ v16bf cat8(v8bf lo, v8bf hi) {
  return __builtin_shufflevector(lo, hi, 0, 1, 2, 3, 4, 5, 6, 7,
                                         8, 9, 10, 11, 12, 13, 14, 15);
}

__launch_bounds__(256)
__global__ void gemm_bf16_wmma(const float* __restrict__ A,
                               const float* __restrict__ Bm,
                               const float* __restrict__ bias,
                               float* __restrict__ C,
                               int M, int N, int K, int flags) {
  __shared__ __bf16 As[2][TILE_M][LDSK];   // [buf][m][k]
  __shared__ __bf16 Bs[2][TILE_N][LDSK];   // [buf][n][k] (transposed B tile)

  const int tid   = threadIdx.x;
  const int lane  = tid & 31;
  const int wave  = tid >> 5;
  const int l16   = lane & 15;
  const int lhalf = lane >> 4;          // 0: lanes 0-15, 1: lanes 16-31
  const int waveM = (wave >> 2) * 64;   // 0 / 64
  const int waveN = (wave & 3) * 32;    // 0 / 32 / 64 / 96
  const int rowBase = blockIdx.y * TILE_M;
  const int colBase = blockIdx.x * TILE_N;

  // per-thread staging slots (4 float4 for A tile, 4 for B tile)
  int rA[4], cA[4];        // A: row 0..127, col(k) 0,4,..,28
  int kB[4], nB[4];        // B: k-row 0..31, n-col 0,4,..,124
#pragma unroll
  for (int i = 0; i < 4; ++i) {
    int slot = tid + i * 256;
    rA[i] = slot >> 3;
    cA[i] = (slot & 7) << 2;
    kB[i] = slot >> 5;
    nB[i] = (slot & 31) << 2;
  }

  v8f acc[4][2];
#pragma unroll
  for (int mi = 0; mi < 4; ++mi)
#pragma unroll
    for (int ni = 0; ni < 2; ++ni)
      acc[mi][ni] = (v8f)(0.0f);

  float4 ra[4], rb[4];

  // issue all 8 global loads for one K-chunk, no intervening waits
  auto load_chunk = [&](int kc) {
#pragma unroll
    for (int i = 0; i < 4; ++i)
      ra[i] = *(const float4*)(A + (size_t)(rowBase + rA[i]) * K + kc + cA[i]);
#pragma unroll
    for (int i = 0; i < 4; ++i) {
      int gn = colBase + nB[i];
      if (gn + 3 < N) {
        rb[i] = *(const float4*)(Bm + (size_t)(kc + kB[i]) * N + gn);
      } else {
        const float* p = Bm + (size_t)(kc + kB[i]) * N;
        rb[i].x = (gn + 0 < N) ? p[gn + 0] : 0.0f;
        rb[i].y = (gn + 1 < N) ? p[gn + 1] : 0.0f;
        rb[i].z = (gn + 2 < N) ? p[gn + 2] : 0.0f;
        rb[i].w = (gn + 3 < N) ? p[gn + 3] : 0.0f;
      }
    }
    // prefetch two chunks ahead (global_prefetch_b8)
    if (kc + 2 * TILE_K < K)
      __builtin_prefetch(A + (size_t)(rowBase + rA[0]) * K +
                             (kc + 2 * TILE_K), 0, 2);
  };

  // convert fp32 -> bf16 and store staged registers into LDS buffer `buf`
  auto store_chunk = [&](int buf) {
#pragma unroll
    for (int i = 0; i < 4; ++i) {
      v4bf b;
      b[0] = (__bf16)ra[i].x; b[1] = (__bf16)ra[i].y;
      b[2] = (__bf16)ra[i].z; b[3] = (__bf16)ra[i].w;
      *(v4bf*)(&As[buf][rA[i]][cA[i]]) = b;
    }
#pragma unroll
    for (int i = 0; i < 4; ++i) {
      Bs[buf][nB[i] + 0][kB[i]] = (__bf16)rb[i].x;
      Bs[buf][nB[i] + 1][kB[i]] = (__bf16)rb[i].y;
      Bs[buf][nB[i] + 2][kB[i]] = (__bf16)rb[i].z;
      Bs[buf][nB[i] + 3][kB[i]] = (__bf16)rb[i].w;
    }
  };

  load_chunk(0);
  int buf = 0;
  for (int kc = 0; kc < K; kc += TILE_K) {
    store_chunk(buf);
    __syncthreads();
    if (kc + TILE_K < K)
      load_chunk(kc + TILE_K);   // in flight during the WMMA block below

    // ---- fragments from LDS[buf] ----
    // A 16x32 bf16: lanes 0-15 row=l16 hold K{0-7,16-23}; lanes 16-31 K{8-15,24-31}
    const int kbA = lhalf * 8;
    v16bf afr[4];
#pragma unroll
    for (int mi = 0; mi < 4; ++mi) {
      const __bf16* p = &As[buf][waveM + mi * 16 + l16][0];
      afr[mi] = cat8(*(const v8bf*)(p + kbA), *(const v8bf*)(p + kbA + 16));
    }
    // B 32x16 bf16: lanes 0-15 col=l16 hold K{0-7,16-23}; lanes 16-31 K{8-15,24-31}
    const int kbB = lhalf * 8;
    v16bf bfr[2];
#pragma unroll
    for (int ni = 0; ni < 2; ++ni) {
      const __bf16* p = &Bs[buf][waveN + ni * 16 + l16][0];
      bfr[ni] = cat8(*(const v8bf*)(p + kbB), *(const v8bf*)(p + kbB + 16));
    }

    // ---- 8 WMMAs: D = A x B + C (f32 accum) ----
#pragma unroll
    for (int mi = 0; mi < 4; ++mi)
#pragma unroll
      for (int ni = 0; ni < 2; ++ni)
        acc[mi][ni] = __builtin_amdgcn_wmma_f32_16x16x32_bf16(
            false, afr[mi], false, bfr[ni], (short)0, acc[mi][ni],
            false, false);

    buf ^= 1;
  }

  // ---- epilogue: C/D layout m = v + 8*lhalf, n = l16 ----
#pragma unroll
  for (int ni = 0; ni < 2; ++ni) {
    int gcol = colBase + waveN + ni * 16 + l16;
    if (gcol >= N) continue;
    float bv = bias ? bias[gcol] : 0.0f;
#pragma unroll
    for (int mi = 0; mi < 4; ++mi) {
      int grow = rowBase + waveM + mi * 16 + lhalf * 8;
#pragma unroll
      for (int v = 0; v < 8; ++v) {
        size_t idx = (size_t)(grow + v) * N + gcol;
        float val = acc[mi][ni][v] + bv;
        if (flags & FLAG_ACC)  val += C[idx];
        if (flags & FLAG_RELU) val = fmaxf(val, 0.0f);
        C[idx] = val;
      }
    }
  }
}

// ---------------------------------------------------------------------------

extern "C" void kernel_launch(void* const* d_in, const int* in_sizes, int n_in,
                              void* d_out, int out_size, void* d_ws,
                              size_t ws_size, hipStream_t stream) {
  constexpr int B = 2048, D = 1024, H = 1024, E = 32, EBM = 128;

  const float* x   = (const float*)d_in[0];
  const float* We  = (const float*)d_in[1];
  const float* be  = (const float*)d_in[2];
  const float* W1  = (const float*)d_in[3];
  const float* b1  = (const float*)d_in[4];
  const float* W2  = (const float*)d_in[5];
  const float* b2  = (const float*)d_in[6];
  const float* Wq1 = (const float*)d_in[7];
  const float* bq1 = (const float*)d_in[8];
  const float* Wq2 = (const float*)d_in[9];
  const float* bq2 = (const float*)d_in[10];
  const float* Wf  = (const float*)d_in[11];
  const float* bf  = (const float*)d_in[12];
  const float* Wh  = (const float*)d_in[13];
  const float* bh  = (const float*)d_in[14];

  float* logits   = (float*)d_out;                  // [B, D]
  float* energies = logits + (size_t)B * D;         // [B, E]

  float* ws    = (float*)d_ws;
  float* x_emb = ws;                                // [B, H]
  float* q1    = x_emb + (size_t)B * H;             // [B, EBM]
  float* h     = q1    + (size_t)B * EBM;           // [B, H]
  float* eo    = h     + (size_t)B * H;             // [B, H]
  float* fused = eo    + (size_t)B * H;             // [B, H]

  auto gemm = [&](const float* A, const float* Bm, const float* bias,
                  float* C, int M, int N, int K, int flags) {
    dim3 grid((N + TILE_N - 1) / TILE_N, M / TILE_M);
    gemm_bf16_wmma<<<grid, dim3(256), 0, stream>>>(A, Bm, bias, C, M, N, K,
                                                   flags);
  };

  // x_emb = x @ We + be
  gemm(x, We, be, x_emb, B, H, D, 0);

  // energies = relu(x_emb @ Wq1 + bq1) @ Wq2 + bq2
  gemm(x_emb, Wq1, bq1, q1, B, EBM, H, FLAG_RELU);
  gemm(q1, Wq2, bq2, energies, B, E, EBM, 0);

  // per-expert: h=relu(x_emb@W1[e]+b1[e]); eo=h@W2[e]+b2[e];
  // fused (+)= eo @ Wf[e*H:(e+1)*H, :]   (bias bf added on e==0)
  for (int e = 0; e < E; ++e) {
    gemm(x_emb, W1 + (size_t)e * H * H, b1 + (size_t)e * H, h, B, H, H,
         FLAG_RELU);
    gemm(h, W2 + (size_t)e * H * H, b2 + (size_t)e * H, eo, B, H, H, 0);
    gemm(eo, Wf + (size_t)e * H * H, (e == 0) ? bf : nullptr, fused, B, H, H,
         (e == 0) ? 0 : FLAG_ACC);
  }

  // logits = fused @ Wh + bh
  gemm(fused, Wh, bh, logits, B, D, H, 0);
}
